// GAT_91122026152131
// MI455X (gfx1250) — matrix operand order repaired and checked by reading, not compiled, not run on hardware
//
#include <hip/hip_runtime.h>
#include <hip/hip_bf16.h>

// ---------------------------------------------------------------------------
// Fused dense-GAT for MI455X (gfx1250, wave32).
//
// logit[i,j] = leakyrelu(s_i + d_j); lrelu monotone => rowmax_i =
// lrelu(s_i + max_j d_j)  -> no online max; e = exp(logit - rowmax) in (0,1]
// -> exact fit for f16 WMMA A-matrix with f32 accumulation.
//
// Layer-0 trick: B columns 8..15 are unused (f_out=8). Column 8 is baked to
// 1.0 in the transposed f16 feature store, so the same v_wmma that aggregates
// sum(e * h') also produces D[:,8] = sum(e) — the softmax denominator — with
// numerator-consistent f16 rounding and f32 accumulation, for free.
//
// Transposed f16 feature store hpT[col][n]: each lane's B fragment is 16
// contiguous halves -> one 32B vector load (2x global_load_b128) per WMMA.
// d is loaded as float4s. The attention matrix never touches HBM.
// ---------------------------------------------------------------------------

#define GAT_N 4096

typedef __attribute__((ext_vector_type(16))) _Float16 v16h;
typedef __attribute__((ext_vector_type(8)))  float    v8f;

// ---------------------------------------------------------------------------
// prep0: hb = BN0(x);  hp0T[h][f][row] = (hb . w0[h][:,f]) as f16 (transposed
// for vectorized WMMA-B loads); col 8 = 1.0 (denominator column), 9..15 = 0.
// s0/d0 = attention source/dest scores.  One thread per (head,row).
// ---------------------------------------------------------------------------
__global__ __launch_bounds__(256) void gat_prep0(
    const float* __restrict__ x,
    const float* __restrict__ g, const float* __restrict__ be,
    const float* __restrict__ mu, const float* __restrict__ var,
    const float* __restrict__ w0,
    const float* __restrict__ as0, const float* __restrict__ ad0,
    _Float16* __restrict__ hp0T, float* __restrict__ s0, float* __restrict__ d0)
{
    int idx  = blockIdx.x * 256 + threadIdx.x;       // 8*4096 threads
    int head = idx >> 12;
    int row  = idx & (GAT_N - 1);

    float hb[32];
    #pragma unroll
    for (int i = 0; i < 32; ++i)
        hb[i] = (x[row * 32 + i] - mu[i]) * g[i] * rsqrtf(var[i] + 1e-5f) + be[i];

    const float* w   = w0 + head * 32 * 8;
    _Float16*    hpT = hp0T + (size_t)head * 16 * GAT_N;
    float s = 0.f, d = 0.f;
    #pragma unroll
    for (int f = 0; f < 8; ++f) {
        float acc = 0.f;
        #pragma unroll
        for (int i = 0; i < 32; ++i) acc += hb[i] * w[i * 8 + f];
        hpT[(size_t)f * GAT_N + row] = (_Float16)acc;
        s += acc * as0[head * 8 + f];
        d += acc * ad0[head * 8 + f];
    }
    hpT[(size_t)8 * GAT_N + row] = (_Float16)1.0f;   // denominator column
    #pragma unroll
    for (int f = 9; f < 16; ++f)
        hpT[(size_t)f * GAT_N + row] = (_Float16)0.0f;

    s0[head * GAT_N + row] = s;
    d0[head * GAT_N + row] = d;
}

// ---------------------------------------------------------------------------
// per-head max over d  (one wave32 per head)
// ---------------------------------------------------------------------------
__global__ void gat_dmax(const float* __restrict__ d, float* __restrict__ dmax,
                         int nheads, int n)
{
    int head = threadIdx.x >> 5;
    int lane = threadIdx.x & 31;
    if (head >= nheads) return;
    float m = -3.4e38f;
    for (int j = lane; j < n; j += 32) m = fmaxf(m, d[head * n + j]);
    #pragma unroll
    for (int o = 16; o; o >>= 1) m = fmaxf(m, __shfl_xor(m, o, 32));
    if (lane == 0) dmax[head] = m;
}

// ---------------------------------------------------------------------------
// attn0: per wave: 16 rows of one head; sweep j in steps of 32.
//   A(16x32 f16) = exp(lrelu(s_i+d_j) - rowmax_i)   (contiguous k = 16h+s, same as B)
//   B(32x16 f16) = hp0T cols 0..7 | ones | zeros     (one v16h load)
//   D += A@B ; D[:,8] = row sums (free denominator)
// out = D[:,0:8]/D[:,8] + b0, ELU, written into concat column h*8+f of h1.
// ---------------------------------------------------------------------------
__global__ __launch_bounds__(256) void gat_attn0(
    const float* __restrict__ s0, const float* __restrict__ d0,
    const float* __restrict__ dmax0, const _Float16* __restrict__ hp0T,
    const float* __restrict__ b0, float* __restrict__ h1)
{
    const int lane = threadIdx.x & 31;
    const int wave = threadIdx.x >> 5;
    const int t    = blockIdx.x * 8 + wave;          // 2048 tiles
    const int head = t >> 8;                         // 0..7
    const int i0   = (t & 255) * 16;

    const float*    sh  = s0 + head * GAT_N;
    const float*    dh  = d0 + head * GAT_N;
    const _Float16* hpT = hp0T + (size_t)head * 16 * GAT_N;

    const int m  = lane & 15;              // A row / B col / D col
    const int kg = (lane >> 4) << 3;       // unused; A loads also use kb
    const int kb = (lane >> 4) << 4;       // B k-group: 0 or 16

    const float sm = sh[i0 + m];
    float mm = sm + dmax0[head];
    mm = mm >= 0.f ? mm : 0.2f * mm;       // rowmax = lrelu(s_i + dmax)

    const _Float16* bptr = hpT + (size_t)m * GAT_N + kb;   // contiguous 16 halves
    v8f acc = {};

    for (int j0 = 0; j0 < GAT_N; j0 += 32) {
        // d for this lane's k-set: two contiguous 8-float runs -> 4x b128
        const float4* p0 = (const float4*)(dh + j0 + kb);
        const float4* p1 = (const float4*)(dh + j0 + kb + 8);
        float4 la = p0[0], lb = p0[1], ha = p1[0], hc = p1[1];
        float dv[16] = { la.x, la.y, la.z, la.w, lb.x, lb.y, lb.z, lb.w,
                         ha.x, ha.y, ha.z, ha.w, hc.x, hc.y, hc.z, hc.w };
        v16h a;
        #pragma unroll
        for (int idx = 0; idx < 16; ++idx) {
            float lg = sm + dv[idx];
            lg = lg >= 0.f ? lg : 0.2f * lg;
            a[idx] = (_Float16)__expf(lg - mm);          // in (0,1]
        }
        v16h b = *(const v16h*)(bptr + j0);              // 2x global_load_b128
        acc = __builtin_amdgcn_wmma_f32_16x16x32_f16(
                  false, a, false, b, (short)0, acc, false, false);
    }

    // denominator: D column 8 lives in lane 8 (rows 0-7) / lane 24 (rows 8-15)
    const int   srcl = 8 + (lane & 16);
    const float bias = (m < 8) ? b0[m] : 0.f;
    #pragma unroll
    for (int r = 0; r < 8; ++r) {
        int   M  = r + ((lane >> 4) << 3);               // D row for this VGPR
        float rs = __shfl(acc[r], srcl, 32);             // row sum from col 8
        if (m < 8) {
            float v = acc[r] / rs + bias;
            v = v > 0.f ? v : (__expf(v) - 1.f);         // ELU
            h1[(size_t)(i0 + M) * 64 + head * 8 + m] = v;
        }
    }
}

// ---------------------------------------------------------------------------
// prep1: hb = BN1(h1); hp1T[f][row] = hb . w1[:,f] (transposed f16); s1,d1.
// ---------------------------------------------------------------------------
__global__ __launch_bounds__(256) void gat_prep1(
    const float* __restrict__ h1,
    const float* __restrict__ g, const float* __restrict__ be,
    const float* __restrict__ mu, const float* __restrict__ var,
    const float* __restrict__ w1,
    const float* __restrict__ as1, const float* __restrict__ ad1,
    _Float16* __restrict__ hp1T, float* __restrict__ s1, float* __restrict__ d1)
{
    int row = blockIdx.x * 256 + threadIdx.x;            // 4096 threads
    float hb[64];
    #pragma unroll
    for (int i = 0; i < 64; ++i)
        hb[i] = (h1[row * 64 + i] - mu[i]) * g[i] * rsqrtf(var[i] + 1e-5f) + be[i];

    float s = 0.f, d = 0.f;
    #pragma unroll 4
    for (int f = 0; f < 32; ++f) {
        float acc = 0.f;
        #pragma unroll
        for (int i = 0; i < 64; ++i) acc += hb[i] * w1[i * 32 + f];
        hp1T[(size_t)f * GAT_N + row] = (_Float16)acc;
        s += acc * as1[f];
        d += acc * ad1[f];
    }
    s1[row] = s;
    d1[row] = d;
}

// ---------------------------------------------------------------------------
// attn1: F=32 -> two B chunks (cols n, n+16), two WMMAs/tile; VALU row sums.
// ---------------------------------------------------------------------------
__global__ __launch_bounds__(256) void gat_attn1(
    const float* __restrict__ s1, const float* __restrict__ d1,
    const float* __restrict__ dmax1, const _Float16* __restrict__ hp1T,
    const float* __restrict__ b1, float* __restrict__ out)
{
    const int lane = threadIdx.x & 31;
    const int wave = threadIdx.x >> 5;
    const int i0   = (blockIdx.x * 8 + wave) * 16;       // 256 tiles

    const int m  = lane & 15;
    const int kg = (lane >> 4) << 3;
    const int kb = (lane >> 4) << 4;

    const float sm = s1[i0 + m];
    float mm = sm + dmax1[0];
    mm = mm >= 0.f ? mm : 0.2f * mm;

    const _Float16* bpA = hp1T + (size_t)m * GAT_N + kb;
    const _Float16* bpB = hp1T + (size_t)(m + 16) * GAT_N + kb;

    v8f acc0 = {}, acc1 = {};
    float rsum = 0.f;

    for (int j0 = 0; j0 < GAT_N; j0 += 32) {
        const float4* p0 = (const float4*)(d1 + j0 + kb);
        const float4* p1 = (const float4*)(d1 + j0 + kb + 8);
        float4 la = p0[0], lb = p0[1], ha = p1[0], hc = p1[1];
        float dv[16] = { la.x, la.y, la.z, la.w, lb.x, lb.y, lb.z, lb.w,
                         ha.x, ha.y, ha.z, ha.w, hc.x, hc.y, hc.z, hc.w };
        v16h a;
        #pragma unroll
        for (int idx = 0; idx < 16; ++idx) {
            float lg = sm + dv[idx];
            lg = lg >= 0.f ? lg : 0.2f * lg;
            _Float16 eh = (_Float16)__expf(lg - mm);
            a[idx] = eh;
            rsum  += (float)eh;                          // denom matches f16 numer
        }
        v16h bA = *(const v16h*)(bpA + j0);
        v16h bB = *(const v16h*)(bpB + j0);
        acc0 = __builtin_amdgcn_wmma_f32_16x16x32_f16(
                   false, a, false, bA, (short)0, acc0, false, false);
        acc1 = __builtin_amdgcn_wmma_f32_16x16x32_f16(
                   false, a, false, bB, (short)0, acc1, false, false);
    }

    // lane L and L^16 cover complementary halves of row (L&15)'s k-space
    float full = rsum + __shfl_xor(rsum, 16, 32);
    #pragma unroll
    for (int r = 0; r < 8; ++r) {
        int   M  = r + ((lane >> 4) << 3);
        float rs = __shfl(full, M, 32);
        out[(size_t)(i0 + M) * 32 + m]      = acc0[r] / rs + b1[m];
        out[(size_t)(i0 + M) * 32 + m + 16] = acc1[r] / rs + b1[m + 16];
    }
}

// ---------------------------------------------------------------------------
extern "C" void kernel_launch(void* const* d_in, const int* in_sizes, int n_in,
                              void* d_out, int out_size, void* d_ws, size_t ws_size,
                              hipStream_t stream)
{
    (void)in_sizes; (void)n_in; (void)out_size; (void)ws_size;

    const float* x         = (const float*)d_in[0];
    const float* bn0_gamma = (const float*)d_in[1];
    const float* bn0_beta  = (const float*)d_in[2];
    const float* bn0_mean  = (const float*)d_in[3];
    const float* bn0_var   = (const float*)d_in[4];
    const float* bn1_gamma = (const float*)d_in[5];
    const float* bn1_beta  = (const float*)d_in[6];
    const float* bn1_mean  = (const float*)d_in[7];
    const float* bn1_var   = (const float*)d_in[8];
    const float* w0        = (const float*)d_in[9];
    const float* a_src0    = (const float*)d_in[10];
    const float* a_dst0    = (const float*)d_in[11];
    const float* b0        = (const float*)d_in[12];
    const float* w1        = (const float*)d_in[13];
    const float* a_src1    = (const float*)d_in[14];
    const float* a_dst1    = (const float*)d_in[15];
    const float* b1        = (const float*)d_in[16];
    float* out = (float*)d_out;

    char* ws = (char*)d_ws;
    _Float16* hp0T  = (_Float16*)(ws + 0);            // 8*16*4096*2 = 1MB
    float*    s0    = (float*)(ws + 1048576);         // 128KB
    float*    d0    = (float*)(ws + 1179648);         // 128KB
    float*    dmax0 = (float*)(ws + 1310720);         // 256B
    float*    h1    = (float*)(ws + 1310976);         // 4096*64*4 = 1MB
    _Float16* hp1T  = (_Float16*)(ws + 2359552);      // 32*4096*2 = 256KB
    float*    s1    = (float*)(ws + 2621696);         // 16KB
    float*    d1    = (float*)(ws + 2638080);         // 16KB
    float*    dmax1 = (float*)(ws + 2654464);         // 256B

    // layer 0
    gat_prep0<<<128, 256, 0, stream>>>(x, bn0_gamma, bn0_beta, bn0_mean, bn0_var,
                                       w0, a_src0, a_dst0, hp0T, s0, d0);
    gat_dmax<<<1, 256, 0, stream>>>(d0, dmax0, 8, GAT_N);
    gat_attn0<<<256, 256, 0, stream>>>(s0, d0, dmax0, hp0T, b0, h1);
    // layer 1
    gat_prep1<<<16, 256, 0, stream>>>(h1, bn1_gamma, bn1_beta, bn1_mean, bn1_var,
                                      w1, a_src1, a_dst1, hp1T, s1, d1);
    gat_dmax<<<1, 32, 0, stream>>>(d1, dmax1, 1, GAT_N);
    gat_attn1<<<32, 256, 0, stream>>>(s1, d1, dmax1, hp1T, b1, out);
}
